// CoTModule_87419764342875
// MI455X (gfx1250) — compile-verified, matrix operand order checked by reading, not run on hardware
//
#include <hip/hip_runtime.h>
#include <cstddef>
#include <cstdint>

// ---------------- model dimensions ----------------
#define Bsz  32
#define TIN  128
#define Dm   512
#define Hh   8
#define HDim 64
#define DFFm 2048
#define TCn  64
#define VCn  64
#define TQn  65                 // T_C + 1
#define RENC (Bsz*TIN)          // 4096 rows
#define RDEC (Bsz*TQn)          // 2080 rows

// ---------------- WMMA fragment types ----------------
typedef __attribute__((ext_vector_type(16))) __bf16 v16bf;
typedef __attribute__((ext_vector_type(8)))  float  v8f;
struct __align__(16) U4 { unsigned x, y, z, w; };
union FragAB { v16bf v; U4 q[2]; };

// =====================================================
// f32[K][N] -> bf16[N][K] transposed weight conversion
// (one-time per launch; makes GEMM B staging contiguous)
// =====================================================
__global__ void cvt_t_bf16_kernel(const float* __restrict__ s, __bf16* __restrict__ d,
                                  int K, int N) {
  const size_t n = (size_t)K * N;
  size_t i  = (size_t)blockIdx.x * blockDim.x + threadIdx.x;
  size_t st = (size_t)gridDim.x * blockDim.x;
  for (; i < n; i += st) {
    const int k = (int)(i / N);
    const int c = (int)(i % N);
    d[(size_t)c * K + k] = (__bf16)s[i];   // coalesced read, strided one-time write
  }
}

// =====================================================
// GEMM: Out[M,N] = A[M,K](f32) * W[N][K](bf16, N-major) (+ Res)
// block tile 128x64, 4 waves; per wave: 2 A-frags x 4 B-frags = 8 WMMA / K-chunk
// single-LDS-buffer register pipeline (load k+1 while computing k)
// =====================================================
#define GTM 128
#define GTN 64
#define GTK 32
#define LDS_A 40   // padded row stride (bf16): 80B rows, 16B-aligned, bank-spread
#define LDS_B 40

__global__ __launch_bounds__(128)
void gemm_bf16_wmma_kernel(const float* __restrict__ A, const __bf16* __restrict__ BwT,
                           const float* __restrict__ Res, float* __restrict__ Out,
                           int M, int N, int K)
{
  __shared__ __bf16 As[GTM * LDS_A];   // 10 KB
  __shared__ __bf16 Bs[GTN * LDS_B];   //  5 KB
  const int tid  = threadIdx.x;
  const int lane = tid & 31;
  const int wav  = tid >> 5;           // 0..3 -> 32-row M sub-tile
  const int bm = blockIdx.y * GTM;
  const int bn = blockIdx.x * GTN;

  const v8f zero = {0.f,0.f,0.f,0.f,0.f,0.f,0.f,0.f};
  v8f acc[2][4] = {{zero,zero,zero,zero},{zero,zero,zero,zero}};

  // staging assignment: thread -> (row ra & ra+64, k-half) ; B: (n=ra, k-half)
  const int ra = tid >> 1;             // 0..63
  const int kh = (tid & 1) * 16;       // 0 / 16
  const bool ok0 = (bm + ra)      < M;
  const bool ok1 = (bm + ra + 64) < M;
  const float*  a0p = A   + (size_t)(bm + ra)      * K + kh;
  const float*  a1p = A   + (size_t)(bm + ra + 64) * K + kh;
  const __bf16* bp  = BwT + (size_t)(bn + ra)      * K + kh;

  // WMMA lane mapping (ISA 7.12.2, 16-bit operands)
  const int fr = lane & 15;            // M (A) / N (B) index
  const int kb = (lane >> 4) * 8;      // K base of the two 8-element runs

  float4 ar0[4], ar1[4];
  U4 br0, br1;
  const float4 f4z = make_float4(0.f, 0.f, 0.f, 0.f);

  auto loadG = [&](int k0) {
    #pragma unroll
    for (int j = 0; j < 4; ++j) {
      ar0[j] = ok0 ? *(const float4*)(a0p + k0 + 4*j) : f4z;
      ar1[j] = ok1 ? *(const float4*)(a1p + k0 + 4*j) : f4z;
    }
    br0 = *(const U4*)(bp + k0);
    br1 = *(const U4*)(bp + k0 + 8);
  };
  auto storeL = [&]() {
    union Pk { __bf16 h[16]; U4 q[2]; } p;
    #pragma unroll
    for (int j = 0; j < 4; ++j) {
      p.h[4*j+0] = (__bf16)ar0[j].x; p.h[4*j+1] = (__bf16)ar0[j].y;
      p.h[4*j+2] = (__bf16)ar0[j].z; p.h[4*j+3] = (__bf16)ar0[j].w;
    }
    *(U4*)&As[ra*LDS_A + kh]     = p.q[0];
    *(U4*)&As[ra*LDS_A + kh + 8] = p.q[1];
    #pragma unroll
    for (int j = 0; j < 4; ++j) {
      p.h[4*j+0] = (__bf16)ar1[j].x; p.h[4*j+1] = (__bf16)ar1[j].y;
      p.h[4*j+2] = (__bf16)ar1[j].z; p.h[4*j+3] = (__bf16)ar1[j].w;
    }
    *(U4*)&As[(ra+64)*LDS_A + kh]     = p.q[0];
    *(U4*)&As[(ra+64)*LDS_A + kh + 8] = p.q[1];
    *(U4*)&Bs[ra*LDS_B + kh]     = br0;
    *(U4*)&Bs[ra*LDS_B + kh + 8] = br1;
  };

  loadG(0);
  for (int k0 = 0; k0 < K; k0 += GTK) {
    storeL();
    __syncthreads();
    if (k0 + GTK < K) {
      if (ok0 && k0 + 2*GTK < K) __builtin_prefetch(a0p + k0 + 2*GTK, 0, 0);
      loadG(k0 + GTK);                 // overlap next global load with WMMAs below
    }

    FragAB fa0, fa1;
    fa0.q[0] = *(const U4*)&As[(wav*32      + fr) * LDS_A + kb];
    fa0.q[1] = *(const U4*)&As[(wav*32      + fr) * LDS_A + kb + 16];
    fa1.q[0] = *(const U4*)&As[(wav*32 + 16 + fr) * LDS_A + kb];
    fa1.q[1] = *(const U4*)&As[(wav*32 + 16 + fr) * LDS_A + kb + 16];
    #pragma unroll
    for (int ct = 0; ct < 4; ++ct) {
      FragAB fb;
      fb.q[0] = *(const U4*)&Bs[(ct*16 + fr) * LDS_B + kb];
      fb.q[1] = *(const U4*)&Bs[(ct*16 + fr) * LDS_B + kb + 16];
      acc[0][ct] = __builtin_amdgcn_wmma_f32_16x16x32_bf16(
          false, fa0.v, false, fb.v, (short)0, acc[0][ct], false, false);
      acc[1][ct] = __builtin_amdgcn_wmma_f32_16x16x32_bf16(
          false, fa1.v, false, fb.v, (short)0, acc[1][ct], false, false);
    }
    __syncthreads();
  }

  // epilogue: C/D layout M = i + 8*(lane>=16), N = lane&15
  const int mh = (lane >> 4) * 8;
  #pragma unroll
  for (int g = 0; g < 2; ++g) {
    #pragma unroll
    for (int ct = 0; ct < 4; ++ct) {
      const int col = bn + ct*16 + fr;
      #pragma unroll
      for (int i = 0; i < 8; ++i) {
        const int row = bm + wav*32 + g*16 + mh + i;
        if (row < M) {
          size_t o = (size_t)row * N + col;
          float v = acc[g][ct][i];
          if (Res) v += Res[o];
          Out[o] = v;
        }
      }
    }
  }
}

// =====================================================
// LayerNorm: one wave32 per row of 512
// =====================================================
__global__ __launch_bounds__(256)
void layernorm_kernel(const float* __restrict__ X, const float* __restrict__ G,
                      float* __restrict__ Y, int R)
{
  const int lane = threadIdx.x & 31;
  const int wav  = threadIdx.x >> 5;
  const int row  = blockIdx.x * 8 + wav;
  if (row >= R) return;
  const float* x = X + (size_t)row * Dm;
  float4 v[4];
  float s = 0.f, ss = 0.f;
  #pragma unroll
  for (int i = 0; i < 4; ++i) {
    v[i] = *(const float4*)(x + lane*16 + i*4);
    s  += v[i].x + v[i].y + v[i].z + v[i].w;
    ss += v[i].x*v[i].x + v[i].y*v[i].y + v[i].z*v[i].z + v[i].w*v[i].w;
  }
  #pragma unroll
  for (int o = 16; o > 0; o >>= 1) { s += __shfl_xor(s, o, 32); ss += __shfl_xor(ss, o, 32); }
  const float m   = s * (1.f / Dm);
  const float inv = rsqrtf(ss * (1.f / Dm) - m*m + 1e-6f);
  float* y = Y + (size_t)row * Dm;
  #pragma unroll
  for (int i = 0; i < 4; ++i) {
    const int d = lane*16 + i*4;
    const float4 g = *(const float4*)(G + d);
    float4 o4;
    o4.x = (v[i].x - m) * inv * g.x;
    o4.y = (v[i].y - m) * inv * g.y;
    o4.z = (v[i].z - m) * inv * g.z;
    o4.w = (v[i].w - m) * inv * g.w;
    *(float4*)(y + d) = o4;
  }
}

// =====================================================
// Attention: one workgroup per (batch, head); K head tile in LDS
// =====================================================
__global__ __launch_bounds__(256)
void attn_kernel(const float* __restrict__ Q, const float* __restrict__ Kt,
                 const float* __restrict__ Vt, float* __restrict__ O,
                 int Tq, int Tk, int causal)
{
  const int b = blockIdx.x / Hh;
  const int h = blockIdx.x % Hh;
  __shared__ float Ks[TIN * (HDim + 1)];   // +1 pad: stride 65 dwords -> bank-spread
  __shared__ float Sc[8][TIN + 4];
  const int tid = threadIdx.x, lane = tid & 31, wav = tid >> 5;

  for (int idx = tid; idx < Tk * HDim; idx += 256) {
    const int j = idx >> 6, d = idx & 63;
    Ks[j*(HDim+1) + d] = Kt[((size_t)(b*Tk + j))*Dm + h*HDim + d];
  }
  __syncthreads();

  for (int t = wav; t < Tq; t += 8) {
    const float* qp = Q + ((size_t)(b*Tq + t))*Dm + h*HDim;
    float qr[HDim];
    #pragma unroll
    for (int d = 0; d < HDim; ++d) qr[d] = qp[d] * 0.125f;  // 1/sqrt(64)

    float lmax = -3.4e38f;
    for (int j = lane; j < Tk; j += 32) {
      float sacc = 0.f;
      const float* kp = &Ks[j*(HDim+1)];
      #pragma unroll
      for (int d = 0; d < HDim; ++d) sacc += qr[d] * kp[d];
      if (causal && j > t) sacc = -3.4e38f;
      Sc[wav][j] = sacc;
      lmax = fmaxf(lmax, sacc);
    }
    #pragma unroll
    for (int o = 16; o > 0; o >>= 1) lmax = fmaxf(lmax, __shfl_xor(lmax, o, 32));

    float lsum = 0.f;
    for (int j = lane; j < Tk; j += 32) {
      const float e = __expf(Sc[wav][j] - lmax);
      Sc[wav][j] = e;
      lsum += e;
    }
    #pragma unroll
    for (int o = 16; o > 0; o >>= 1) lsum += __shfl_xor(lsum, o, 32);
    const float inv = 1.f / lsum;

    // cross-lane LDS read follows: wave32 lockstep + explicit DS wait
    asm volatile("s_wait_dscnt 0" ::: "memory");

    float o0 = 0.f, o1 = 0.f;
    const int d0 = lane * 2;
    for (int j = 0; j < Tk; ++j) {
      const float p = Sc[wav][j];
      const float* vp = Vt + ((size_t)(b*Tk + j))*Dm + h*HDim;
      o0 += p * vp[d0];
      o1 += p * vp[d0 + 1];
    }
    float* op = O + ((size_t)(b*Tq + t))*Dm + h*HDim;
    op[d0]     = o0 * inv;
    op[d0 + 1] = o1 * inv;
  }
}

// =====================================================
// GELU (tanh approx, matches jax.nn.gelu default)
// =====================================================
__global__ void gelu_kernel(float* __restrict__ X, size_t n) {
  size_t i  = (size_t)blockIdx.x * blockDim.x + threadIdx.x;
  size_t st = (size_t)gridDim.x * blockDim.x;
  for (; i < n; i += st) {
    const float x = X[i];
    const float c = 0.7978845608028654f * (x + 0.044715f * x * x * x);
    X[i] = 0.5f * x * (1.f + tanhf(c));
  }
}

// =====================================================
// Embedding gather: out[row] = tokE[tok[row]] + posE[row % T]
// =====================================================
__global__ void embed_kernel(const int* __restrict__ toks, const float* __restrict__ tokE,
                             const float* __restrict__ posE, float* __restrict__ out,
                             int R, int T)
{
  const int row = blockIdx.x;
  if (row >= R) return;
  const int id = toks[row];
  const int t  = row % T;
  const float* te = tokE + (size_t)id * Dm;
  const float* pe = posE + (size_t)t  * Dm;
  float* o = out + (size_t)row * Dm;
  for (int d = threadIdx.x; d < Dm; d += blockDim.x) o[d] = te[d] + pe[d];
}

// =====================================================
// cot token buffer init: [:,0] = V_C, rest 0
// =====================================================
__global__ void init_cot_kernel(int* __restrict__ cot) {
  const int i = blockIdx.x * blockDim.x + threadIdx.x;
  if (i < Bsz * TQn) cot[i] = ((i % TQn) == 0) ? VCn : 0;
}

// =====================================================
// per-step argmax (first-max tiebreak) + token update + logits copy-out
// =====================================================
__global__ void argmax_update_kernel(const float* __restrict__ logits, int* __restrict__ cot,
                                     float* __restrict__ outF, int step)
{
  const int b = blockIdx.x;
  const int t = threadIdx.x;         // 0..63 (vocab)
  __shared__ float sv[64];
  __shared__ int   si[64];
  const float v = logits[((size_t)(b*TQn + step)) * VCn + t];
  sv[t] = v; si[t] = t;
  outF[2048 + ((size_t)b * TCn + step) * VCn + t] = v;   // logits (B,T_C,V_C)
  __syncthreads();
  for (int s = 32; s > 0; s >>= 1) {
    if (t < s) {
      if (sv[t+s] > sv[t] || (sv[t+s] == sv[t] && si[t+s] < si[t])) {
        sv[t] = sv[t+s]; si[t] = si[t+s];
      }
    }
    __syncthreads();
  }
  if (t == 0) {
    cot[b*TQn + step + 1] = si[0];
    outF[(size_t)b * TCn + step] = (float)si[0];         // tokens region
  }
}

// =====================================================
// host orchestration
// =====================================================
extern "C" void kernel_launch(void* const* d_in, const int* in_sizes, int n_in,
                              void* d_out, int out_size, void* d_ws, size_t ws_size,
                              hipStream_t stream)
{
  (void)in_sizes; (void)n_in; (void)out_size; (void)ws_size;
  auto F = [&](int i) { return (const float*)d_in[i]; };

  const int*   inputs  = (const int*)d_in[0];
  const float* tokE    = F(1);
  const float* posE    = F(2);
  const float* cotTokE = F(35);
  const float* cotPosE = F(36);
  const float* lnF     = F(89);
  const float* headW   = F(90);

  const float *eLn1[4], *eWq[4], *eWk[4], *eWv[4], *eWo[4], *eLn2[4], *eW1[4], *eW2[4];
  for (int l = 0; l < 4; ++l) {
    const int b0 = 3 + 8*l;
    eLn1[l]=F(b0+0); eWq[l]=F(b0+1); eWk[l]=F(b0+2); eWv[l]=F(b0+3);
    eWo[l]=F(b0+4);  eLn2[l]=F(b0+5); eW1[l]=F(b0+6); eW2[l]=F(b0+7);
  }
  const float *xLn1[4], *xWq[4], *xWk[4], *xWv[4], *xWo[4], *xLn2[4],
              *xCWq[4], *xCWk[4], *xCWv[4], *xCWo[4], *xLn3[4], *xW1[4], *xW2[4];
  for (int l = 0; l < 4; ++l) {
    const int b0 = 37 + 13*l;
    xLn1[l]=F(b0+0); xWq[l]=F(b0+1);  xWk[l]=F(b0+2);  xWv[l]=F(b0+3);
    xWo[l]=F(b0+4);  xLn2[l]=F(b0+5); xCWq[l]=F(b0+6); xCWk[l]=F(b0+7);
    xCWv[l]=F(b0+8); xCWo[l]=F(b0+9); xLn3[l]=F(b0+10); xW1[l]=F(b0+11); xW2[l]=F(b0+12);
  }

  // ---- workspace carving (~210 MB) ----
  char*  wsp = (char*)d_ws;
  size_t off = 0;
  auto carve = [&](size_t bytes) -> void* {
    void* p = wsp + off;
    off += (bytes + 255) & ~(size_t)255;
    return p;
  };
  float* ENCX = (float*)carve((size_t)RENC * Dm * 4);
  float* HB   = (float*)carve((size_t)RENC * Dm * 4);
  float* QB   = (float*)carve((size_t)RENC * Dm * 4);
  float* KB   = (float*)carve((size_t)RENC * Dm * 4);
  float* VB   = (float*)carve((size_t)RENC * Dm * 4);
  float* ATT  = (float*)carve((size_t)RENC * Dm * 4);
  float* MLPH = (float*)carve((size_t)RENC * DFFm * 4);
  float* CKb[4], *CVb[4];
  for (int l = 0; l < 4; ++l) {
    CKb[l] = (float*)carve((size_t)RENC * Dm * 4);
    CVb[l] = (float*)carve((size_t)RENC * Dm * 4);
  }
  float* CB  = (float*)carve((size_t)RDEC * Dm * 4);
  float* LG  = (float*)carve((size_t)RDEC * VCn * 4);
  int*   COT = (int*)  carve((size_t)Bsz * TQn * 4);

  auto carveH = [&](size_t elems) { return (__bf16*)carve(elems * 2); };
  const size_t DD = (size_t)Dm*Dm, DDF = (size_t)Dm*DFFm;
  __bf16 *beWq[4], *beWk[4], *beWv[4], *beWo[4], *beW1[4], *beW2[4];
  for (int l = 0; l < 4; ++l) {
    beWq[l]=carveH(DD); beWk[l]=carveH(DD); beWv[l]=carveH(DD); beWo[l]=carveH(DD);
    beW1[l]=carveH(DDF); beW2[l]=carveH(DDF);
  }
  __bf16 *bxWq[4], *bxWk[4], *bxWv[4], *bxWo[4], *bxCWq[4], *bxCWk[4],
         *bxCWv[4], *bxCWo[4], *bxW1[4], *bxW2[4];
  for (int l = 0; l < 4; ++l) {
    bxWq[l]=carveH(DD);  bxWk[l]=carveH(DD);  bxWv[l]=carveH(DD);  bxWo[l]=carveH(DD);
    bxCWq[l]=carveH(DD); bxCWk[l]=carveH(DD); bxCWv[l]=carveH(DD); bxCWo[l]=carveH(DD);
    bxW1[l]=carveH(DDF); bxW2[l]=carveH(DDF);
  }
  __bf16* bHead = carveH((size_t)Dm * VCn);

  // ---- launch helpers ----
  auto cvtT = [&](const float* s, __bf16* d, int K, int N) {
    size_t n = (size_t)K * N;
    int grid = (int)((n + 1023) / 1024); if (grid > 2048) grid = 2048; if (grid < 1) grid = 1;
    cvt_t_bf16_kernel<<<grid, 256, 0, stream>>>(s, d, K, N);
  };
  auto gemm = [&](const float* A, const __bf16* BwT, const float* Res, float* Out,
                  int M, int N, int K) {
    dim3 grid(N / GTN, (M + GTM - 1) / GTM);
    gemm_bf16_wmma_kernel<<<grid, 128, 0, stream>>>(A, BwT, Res, Out, M, N, K);
  };
  auto ln = [&](const float* X, const float* G, float* Y, int R) {
    layernorm_kernel<<<(R + 7) / 8, 256, 0, stream>>>(X, G, Y, R);
  };
  auto attn = [&](const float* Q, const float* K, const float* V, float* O,
                  int Tq, int Tk, int causal) {
    attn_kernel<<<Bsz * Hh, 256, 0, stream>>>(Q, K, V, O, Tq, Tk, causal);
  };
  auto gelu = [&](float* X, size_t n) {
    int grid = (int)((n + 1023) / 1024); if (grid > 4096) grid = 4096;
    gelu_kernel<<<grid, 256, 0, stream>>>(X, n);
  };

  // ---- phase 0: weights f32 -> bf16, transposed to N-major (L2-resident) ----
  for (int l = 0; l < 4; ++l) {
    cvtT(eWq[l], beWq[l], Dm, Dm); cvtT(eWk[l], beWk[l], Dm, Dm);
    cvtT(eWv[l], beWv[l], Dm, Dm); cvtT(eWo[l], beWo[l], Dm, Dm);
    cvtT(eW1[l], beW1[l], Dm, DFFm); cvtT(eW2[l], beW2[l], DFFm, Dm);
  }
  for (int l = 0; l < 4; ++l) {
    cvtT(xWq[l],  bxWq[l],  Dm, Dm); cvtT(xWk[l],  bxWk[l],  Dm, Dm);
    cvtT(xWv[l],  bxWv[l],  Dm, Dm); cvtT(xWo[l],  bxWo[l],  Dm, Dm);
    cvtT(xCWq[l], bxCWq[l], Dm, Dm); cvtT(xCWk[l], bxCWk[l], Dm, Dm);
    cvtT(xCWv[l], bxCWv[l], Dm, Dm); cvtT(xCWo[l], bxCWo[l], Dm, Dm);
    cvtT(xW1[l],  bxW1[l],  Dm, DFFm); cvtT(xW2[l], bxW2[l], DFFm, Dm);
  }
  cvtT(headW, bHead, Dm, VCn);

  // ---- phase 1: encoder ----
  embed_kernel<<<RENC, 128, 0, stream>>>(inputs, tokE, posE, ENCX, RENC, TIN);
  for (int l = 0; l < 4; ++l) {
    ln(ENCX, eLn1[l], HB, RENC);
    gemm(HB, beWq[l], nullptr, QB, RENC, Dm, Dm);
    gemm(HB, beWk[l], nullptr, KB, RENC, Dm, Dm);
    gemm(HB, beWv[l], nullptr, VB, RENC, Dm, Dm);
    attn(QB, KB, VB, ATT, TIN, TIN, 0);
    gemm(ATT, beWo[l], ENCX, ENCX, RENC, Dm, Dm);        // x += attn @ Wo
    ln(ENCX, eLn2[l], HB, RENC);
    gemm(HB, beW1[l], nullptr, MLPH, RENC, DFFm, Dm);
    gelu(MLPH, (size_t)RENC * DFFm);
    gemm(MLPH, beW2[l], ENCX, ENCX, RENC, Dm, DFFm);     // x += mlp
  }

  // ---- phase 2: cross-attn K/V precompute (reused across 64 steps) ----
  for (int l = 0; l < 4; ++l) {
    gemm(ENCX, bxCWk[l], nullptr, CKb[l], RENC, Dm, Dm);
    gemm(ENCX, bxCWv[l], nullptr, CVb[l], RENC, Dm, Dm);
  }

  // ---- phase 3: autoregressive decode ----
  init_cot_kernel<<<(Bsz*TQn + 255)/256, 256, 0, stream>>>(COT);
  float* outF = (float*)d_out;

  for (int step = 0; step < TCn; ++step) {
    embed_kernel<<<RDEC, 128, 0, stream>>>(COT, cotTokE, cotPosE, CB, RDEC, TQn);
    for (int l = 0; l < 4; ++l) {
      // causal self-attention
      ln(CB, xLn1[l], HB, RDEC);
      gemm(HB, bxWq[l], nullptr, QB, RDEC, Dm, Dm);
      gemm(HB, bxWk[l], nullptr, KB, RDEC, Dm, Dm);
      gemm(HB, bxWv[l], nullptr, VB, RDEC, Dm, Dm);
      attn(QB, KB, VB, ATT, TQn, TQn, 1);
      gemm(ATT, bxWo[l], CB, CB, RDEC, Dm, Dm);          // c += self
      // cross-attention (K/V precomputed)
      ln(CB, xLn2[l], HB, RDEC);
      gemm(HB, bxCWq[l], nullptr, QB, RDEC, Dm, Dm);
      attn(QB, CKb[l], CVb[l], ATT, TQn, TIN, 0);
      gemm(ATT, bxCWo[l], CB, CB, RDEC, Dm, Dm);         // c += cross
      // MLP
      ln(CB, xLn3[l], HB, RDEC);
      gemm(HB, bxW1[l], nullptr, MLPH, RDEC, DFFm, Dm);
      gelu(MLPH, (size_t)RDEC * DFFm);
      gemm(MLPH, bxW2[l], CB, CB, RDEC, Dm, DFFm);       // c += mlp
    }
    ln(CB, lnF, HB, RDEC);
    gemm(HB, bHead, nullptr, LG, RDEC, VCn, Dm);
    argmax_update_kernel<<<Bsz, 64, 0, stream>>>(LG, COT, outF, step);
  }
}
